// CLAM_SB_55654186221762
// MI455X (gfx1250) — matrix operand order repaired and checked by reading, not compiled, not run on hardware
//
#include <hip/hip_runtime.h>
#include <hip/hip_bf16.h>

#define NN    20000
#define MPAD  20032   // NN rounded up to 64-row tiles
#define PADR  (MPAD - NN)
#define EE    160000
#define F_IN  512
#define H1    256
#define H2    512
#define D_ATT 256

typedef __attribute__((ext_vector_type(16))) __bf16 v16bf;
typedef __attribute__((ext_vector_type(8)))  __bf16 v8bf;
typedef __attribute__((ext_vector_type(8)))  float  v8f;

// ---------------------------------------------------------------------------
// WMMA fragment loaders (CDNA5 16-bit layouts, CDNA5 ISA §7.12.2)
// A: 16x32 (MxK), row-major source [M,K]:
//   lane l: M = l&15 ; kbase = 8*(l>>4); elems 0..7 -> K=kb+kbase+0..7,
//   elems 8..15 -> K=kb+16+kbase+0..7  => two contiguous 16B loads.
// B: 32x16 (KxN), source pre-transposed [N,K]:
//   lane l: N = l&15 ; same K mapping as A (kb+8*(l>>4)+0..7, +16) => 2x16B.
// ---------------------------------------------------------------------------
__device__ inline v16bf load_a_frag(const __bf16* __restrict__ A, int lda,
                                    int rowBase, int kb, int lane) {
  int m    = rowBase + (lane & 15);
  int koff = (lane >> 4) << 3;
  const __bf16* p = A + (size_t)m * lda + kb + koff;
  v8bf lo = *(const v8bf*)p;
  v8bf hi = *(const v8bf*)(p + 16);
  v16bf r;
#pragma unroll
  for (int i = 0; i < 8; ++i) { r[i] = lo[i]; r[i + 8] = hi[i]; }
  return r;
}

__device__ inline v16bf load_b_frag(const __bf16* __restrict__ Wt, int K,
                                    int colBase, int kb, int lane) {
  int n  = colBase + (lane & 15);
  int ks = kb + ((lane >> 4) << 3);
  const __bf16* p = Wt + (size_t)n * K + ks;
  v8bf lo = *(const v8bf*)p;
  v8bf hi = *(const v8bf*)(p + 16);
  v16bf r;
#pragma unroll
  for (int i = 0; i < 8; ++i) { r[i] = lo[i]; r[i + 8] = hi[i]; }
  return r;
}

__device__ inline v8f wmma_bf16(v16bf a, v16bf b, v8f c) {
  return __builtin_amdgcn_wmma_f32_16x16x32_bf16(false, a, false, b,
                                                 (short)0, c, false, false);
}

// ---------------------------------------------------------------------------
// Fused GEMM, 64x64 register tile per wave (4x4 accumulators):
//   out = act(A1@W1t^T [+ A2@W2t^T] + bias)
// Per K-step: 16 B128 loads feed 16 independent WMMAs (hides XDL latency).
// act: 0 none, 1 tanh, 2 relu. M must be a multiple of 64 (padded).
// ---------------------------------------------------------------------------
__global__ __launch_bounds__(256)
void k_gemm(const __bf16* __restrict__ A1, const __bf16* __restrict__ W1t,
            const __bf16* __restrict__ A2, const __bf16* __restrict__ W2t,
            const float* __restrict__ bias,
            float* __restrict__ outF, __bf16* __restrict__ outB,
            int M, int K, int NOUT, int act) {
  int lane = threadIdx.x & 31;
  int wave = threadIdx.x >> 5;
  int nGrpN = NOUT >> 6;                 // 64-col groups
  int tile  = blockIdx.x * 8 + wave;
  int total = (M >> 6) * nGrpN;
  if (tile >= total) return;
  int rt = tile / nGrpN, ct = tile - rt * nGrpN;
  int rowBase = rt << 6, colBase = ct << 6;

  v8f acc[4][4];
#pragma unroll
  for (int i = 0; i < 4; ++i)
#pragma unroll
    for (int j = 0; j < 4; ++j) acc[i][j] = (v8f){};

  for (int kb = 0; kb < K; kb += 32) {
    v16bf a[4], b[4];
#pragma unroll
    for (int i = 0; i < 4; ++i)
      a[i] = load_a_frag(A1, K, rowBase + (i << 4), kb, lane);
#pragma unroll
    for (int j = 0; j < 4; ++j)
      b[j] = load_b_frag(W1t, K, colBase + (j << 4), kb, lane);
#pragma unroll
    for (int i = 0; i < 4; ++i)
#pragma unroll
      for (int j = 0; j < 4; ++j) acc[i][j] = wmma_bf16(a[i], b[j], acc[i][j]);
  }
  if (A2) {
    for (int kb = 0; kb < K; kb += 32) {
      v16bf a[4], b[4];
#pragma unroll
      for (int i = 0; i < 4; ++i)
        a[i] = load_a_frag(A2, K, rowBase + (i << 4), kb, lane);
#pragma unroll
      for (int j = 0; j < 4; ++j)
        b[j] = load_b_frag(W2t, K, colBase + (j << 4), kb, lane);
#pragma unroll
      for (int i = 0; i < 4; ++i)
#pragma unroll
        for (int j = 0; j < 4; ++j)
          acc[i][j] = wmma_bf16(a[i], b[j], acc[i][j]);
    }
  }

  int rOff = (lane >> 4) << 3;           // C layout: row = base + r + 8*(l>>4)
#pragma unroll
  for (int j = 0; j < 4; ++j) {
    int col = colBase + (j << 4) + (lane & 15);
    float bv = bias ? bias[col] : 0.f;
#pragma unroll
    for (int i = 0; i < 4; ++i) {
#pragma unroll
      for (int r = 0; r < 8; ++r) {
        int row = rowBase + (i << 4) + rOff + r;
        float v = acc[i][j][r] + bv;
        if (act == 1) v = tanhf(v);
        else if (act == 2) v = fmaxf(v, 0.f);
        size_t idx = (size_t)row * NOUT + col;
        if (outF) outF[idx] = v;
        if (outB) outB[idx] = (__bf16)v;
      }
    }
  }
}

// ---------------------------------------------------------------------------
// Gated attention, 64x16 tile per wave: a=tanh(H@Wa+ba), b=sig(H@Wb+bb),
// per-row sum of a*b*Wc[col] atomically added into Aatt[row]. K=512.
// A-frags reused across both weight matrices: 8 wmma per 12 B128 loads.
// ---------------------------------------------------------------------------
__global__ __launch_bounds__(256)
void k_attn(const __bf16* __restrict__ H, const __bf16* __restrict__ Wat,
            const __bf16* __restrict__ Wbt, const float* __restrict__ ba,
            const float* __restrict__ bb, const float* __restrict__ Wc,
            float* __restrict__ Aatt, int M) {
  const int K = H2, NOUT = D_ATT;
  int lane = threadIdx.x & 31;
  int wave = threadIdx.x >> 5;
  int nTilesN = NOUT >> 4;               // 16 col tiles
  int tile = blockIdx.x * 8 + wave;
  int total = (M >> 6) * nTilesN;
  if (tile >= total) return;
  int rt = tile / nTilesN, ct = tile - rt * nTilesN;
  int rowBase = rt << 6, colBase = ct << 4;

  v8f acca[4], accb[4];
#pragma unroll
  for (int i = 0; i < 4; ++i) { acca[i] = (v8f){}; accb[i] = (v8f){}; }

  for (int kb = 0; kb < K; kb += 32) {
    v16bf a[4];
#pragma unroll
    for (int i = 0; i < 4; ++i)
      a[i] = load_a_frag(H, K, rowBase + (i << 4), kb, lane);
    v16bf wa = load_b_frag(Wat, K, colBase, kb, lane);
    v16bf wb = load_b_frag(Wbt, K, colBase, kb, lane);
#pragma unroll
    for (int i = 0; i < 4; ++i) {
      acca[i] = wmma_bf16(a[i], wa, acca[i]);
      accb[i] = wmma_bf16(a[i], wb, accb[i]);
    }
  }

  int col = colBase + (lane & 15);
  float bav = ba[col], bbv = bb[col], wcv = Wc[col];
  int rOff = (lane >> 4) << 3;
#pragma unroll
  for (int i = 0; i < 4; ++i) {
#pragma unroll
    for (int r = 0; r < 8; ++r) {
      float av = tanhf(acca[i][r] + bav);
      float sv = 1.f / (1.f + expf(-(accb[i][r] + bbv)));
      float p = av * sv * wcv;
      p += __shfl_xor(p, 1, 32);
      p += __shfl_xor(p, 2, 32);
      p += __shfl_xor(p, 4, 32);
      p += __shfl_xor(p, 8, 32);
      if ((lane & 15) == 0)
        atomicAdd(&Aatt[rowBase + (i << 4) + rOff + r], p);
    }
  }
}

// --------------------------- support kernels -------------------------------
__global__ void k_zero(float* __restrict__ p, int n) {
  int i = blockIdx.x * blockDim.x + threadIdx.x;
  if (i < n) p[i] = 0.f;
}

__global__ void k_f32_to_bf16(const float* __restrict__ in,
                              __bf16* __restrict__ out, int n) {
  int i = blockIdx.x * blockDim.x + threadIdx.x;
  if (i < n) out[i] = (__bf16)in[i];
}

// W [K, NOUT] f32 row-major -> Wt [NOUT, K] bf16
__global__ void k_transpose_bf(const float* __restrict__ W,
                               __bf16* __restrict__ Wt, int K, int NOUT) {
  int i = blockIdx.x * blockDim.x + threadIdx.x;
  if (i >= K * NOUT) return;
  int k = i / NOUT, n = i - k * NOUT;
  Wt[(size_t)n * K + k] = (__bf16)W[i];
}

__global__ void k_deg(const int* __restrict__ dst, float* __restrict__ deg,
                      int ne) {
  int i = blockIdx.x * blockDim.x + threadIdx.x;
  if (i < ne) atomicAdd(&deg[dst[i]], 1.f);
}

// scatter-add X[src[e], :] into agg[dst[e], :], float4 granularity
__global__ void k_scatter4(const float* __restrict__ X,
                           const int* __restrict__ src,
                           const int* __restrict__ dst,
                           float* __restrict__ agg, int ne, int D4) {
  int gid = blockIdx.x * blockDim.x + threadIdx.x;
  if (gid >= ne * D4) return;
  int e = gid / D4, c = gid - e * D4;
  int s = src[e], d = dst[e];
  float4 v = ((const float4*)X)[(size_t)s * D4 + c];
  float* o = agg + (((size_t)d * D4 + c) << 2);
  atomicAdd(o + 0, v.x);
  atomicAdd(o + 1, v.y);
  atomicAdd(o + 2, v.z);
  atomicAdd(o + 3, v.w);
}

__global__ void k_mean_bf(const float* __restrict__ agg,
                          const float* __restrict__ deg,
                          __bf16* __restrict__ out, int total, int D) {
  int i = blockIdx.x * blockDim.x + threadIdx.x;
  if (i >= total) return;
  int row = i / D;
  out[i] = (__bf16)(agg[i] / fmaxf(deg[row], 1.f));
}

// softmax over Aatt+bc; writes A_raw to out and weights to wbuf
__global__ void k_softmax(const float* __restrict__ Aatt,
                          const float* __restrict__ bc,
                          float* __restrict__ Araw, float* __restrict__ wbuf,
                          int M) {
  __shared__ float sred[1024];
  int t = threadIdx.x, nt = blockDim.x;
  float bcv = bc[0];
  float mx = -3.4e38f;
  for (int i = t; i < M; i += nt) mx = fmaxf(mx, Aatt[i] + bcv);
  sred[t] = mx; __syncthreads();
  for (int s = nt >> 1; s > 0; s >>= 1) {
    if (t < s) sred[t] = fmaxf(sred[t], sred[t + s]);
    __syncthreads();
  }
  float gmax = sred[0]; __syncthreads();
  float sm = 0.f;
  for (int i = t; i < M; i += nt) sm += expf(Aatt[i] + bcv - gmax);
  sred[t] = sm; __syncthreads();
  for (int s = nt >> 1; s > 0; s >>= 1) {
    if (t < s) sred[t] += sred[t + s];
    __syncthreads();
  }
  float inv = 1.f / sred[0];
  for (int i = t; i < M; i += nt) {
    float Ai = Aatt[i] + bcv;
    Araw[i] = Ai;
    wbuf[i] = expf(Ai - gmax) * inv;
  }
}

// M[512] += sum_i w[i] * h[i, :], block handles 200 rows, 256 threads x 2 dims
__global__ __launch_bounds__(256)
void k_pool(const float* __restrict__ h, const float* __restrict__ wbuf,
            float* __restrict__ Mvec, int M) {
  int t = threadIdx.x;
  int r0 = blockIdx.x * 200;
  int r1 = min(r0 + 200, M);
  float a0 = 0.f, a1 = 0.f;
  for (int i = r0; i < r1; ++i) {
    float w = wbuf[i];
    a0 += w * h[(size_t)i * H2 + t];
    a1 += w * h[(size_t)i * H2 + t + 256];
  }
  atomicAdd(&Mvec[t], a0);
  atomicAdd(&Mvec[t + 256], a1);
}

__global__ void k_final(const float* __restrict__ Mvec,
                        const float* __restrict__ Wcls,
                        const float* __restrict__ bcls,
                        float* __restrict__ out) {
  __shared__ float s0[256], s1[256];
  int t = threadIdx.x;
  float p0 = 0.f, p1 = 0.f;
  for (int d = t; d < H2; d += 256) {
    float m = Mvec[d];
    p0 += m * Wcls[d * 2 + 0];
    p1 += m * Wcls[d * 2 + 1];
  }
  s0[t] = p0; s1[t] = p1; __syncthreads();
  for (int s = 128; s > 0; s >>= 1) {
    if (t < s) { s0[t] += s0[t + s]; s1[t] += s1[t + s]; }
    __syncthreads();
  }
  if (t == 0) {
    float l0 = s0[0] + bcls[0], l1 = s1[0] + bcls[1];
    out[0] = l0; out[1] = l1;
    float mx = fmaxf(l0, l1);
    float e0 = expf(l0 - mx), e1 = expf(l1 - mx);
    float ss = e0 + e1;
    out[2] = e0 / ss; out[3] = e1 / ss;
    out[4] = (l1 > l0) ? 1.f : 0.f;
  }
}

// ---------------------------------------------------------------------------
extern "C" void kernel_launch(void* const* d_in, const int* in_sizes, int n_in,
                              void* d_out, int out_size, void* d_ws,
                              size_t ws_size, hipStream_t stream) {
  const float* x    = (const float*)d_in[0];
  const int*   edge = (const int*)d_in[2];
  const int*   src  = edge;
  const int*   dstp = edge + EE;
  const float* W1l = (const float*)d_in[4];
  const float* b1  = (const float*)d_in[5];
  const float* W1r = (const float*)d_in[6];
  const float* W2l = (const float*)d_in[7];
  const float* b2  = (const float*)d_in[8];
  const float* W2r = (const float*)d_in[9];
  const float* Wfc = (const float*)d_in[10];
  const float* bfc = (const float*)d_in[11];
  const float* Wa  = (const float*)d_in[12];
  const float* ba  = (const float*)d_in[13];
  const float* Wb  = (const float*)d_in[14];
  const float* bb  = (const float*)d_in[15];
  const float* Wc  = (const float*)d_in[16];
  const float* bc  = (const float*)d_in[17];
  const float* Wcls = (const float*)d_in[18];
  const float* bcls = (const float*)d_in[19];
  float* out = (float*)d_out;

  // ---- workspace layout (256B aligned); activations padded to MPAD rows ----
  char* ws = (char*)d_ws;
  size_t off = 0;
  auto alloc = [&](size_t bytes) {
    size_t o = off;
    off += (bytes + 255) & ~(size_t)255;
    return (void*)(ws + o);
  };
  float*  deg   = (float*)alloc((size_t)NN * 4);
  float*  agg1  = (float*)alloc((size_t)MPAD * F_IN * 4);   // later: h_f32
  __bf16* xbf   = (__bf16*)alloc((size_t)MPAD * F_IN * 2);  // later: h2_bf
  __bf16* m1bf  = (__bf16*)alloc((size_t)MPAD * F_IN * 2);  // later: h_bf
  float*  h1f   = (float*)alloc((size_t)MPAD * H1 * 4);
  __bf16* h1bf  = (__bf16*)alloc((size_t)MPAD * H1 * 2);
  float*  agg2  = (float*)alloc((size_t)NN * H1 * 4);
  __bf16* m2bf  = (__bf16*)alloc((size_t)MPAD * H1 * 2);
  float*  Aatt  = (float*)alloc((size_t)MPAD * 4);
  float*  wbuf  = (float*)alloc((size_t)NN * 4);
  float*  Mvec  = (float*)alloc(512 * 4);
  __bf16* W1lt = (__bf16*)alloc((size_t)F_IN * H1 * 2);
  __bf16* W1rt = (__bf16*)alloc((size_t)F_IN * H1 * 2);
  __bf16* W2lt = (__bf16*)alloc((size_t)H1 * H2 * 2);
  __bf16* W2rt = (__bf16*)alloc((size_t)H1 * H2 * 2);
  __bf16* Wfct = (__bf16*)alloc((size_t)H2 * H2 * 2);
  __bf16* Wat  = (__bf16*)alloc((size_t)H2 * D_ATT * 2);
  __bf16* Wbt  = (__bf16*)alloc((size_t)H2 * D_ATT * 2);
  float*  hF  = agg1;  // post-fc features f32 (reuse)
  __bf16* h2bf = xbf;  // layer-2 output bf16 (reuse)
  __bf16* hbf  = m1bf; // post-fc features bf16 (reuse)

  auto cdiv = [](int a, int b) { return (a + b - 1) / b; };

  // ---- 0. zero accumulators + pad tails of GEMM inputs ----
  k_zero<<<cdiv(NN, 256), 256, 0, stream>>>(deg, NN);
  k_zero<<<cdiv(NN * F_IN, 256), 256, 0, stream>>>(agg1, NN * F_IN);
  k_zero<<<cdiv(NN * H1, 256), 256, 0, stream>>>(agg2, NN * H1);
  k_zero<<<cdiv(MPAD, 256), 256, 0, stream>>>(Aatt, MPAD);
  k_zero<<<cdiv(512, 256), 256, 0, stream>>>(Mvec, 512);
  k_zero<<<cdiv(PADR * F_IN / 2, 256), 256, 0, stream>>>(
      (float*)(xbf + (size_t)NN * F_IN), PADR * F_IN / 2);
  k_zero<<<cdiv(PADR * F_IN / 2, 256), 256, 0, stream>>>(
      (float*)(m1bf + (size_t)NN * F_IN), PADR * F_IN / 2);
  k_zero<<<cdiv(PADR * H1 / 2, 256), 256, 0, stream>>>(
      (float*)(m2bf + (size_t)NN * H1), PADR * H1 / 2);

  // ---- 1. weight transpose/convert to bf16 [Nout, K] ----
  k_transpose_bf<<<cdiv(F_IN * H1, 256), 256, 0, stream>>>(W1l, W1lt, F_IN, H1);
  k_transpose_bf<<<cdiv(F_IN * H1, 256), 256, 0, stream>>>(W1r, W1rt, F_IN, H1);
  k_transpose_bf<<<cdiv(H1 * H2, 256), 256, 0, stream>>>(W2l, W2lt, H1, H2);
  k_transpose_bf<<<cdiv(H1 * H2, 256), 256, 0, stream>>>(W2r, W2rt, H1, H2);
  k_transpose_bf<<<cdiv(H2 * H2, 256), 256, 0, stream>>>(Wfc, Wfct, H2, H2);
  k_transpose_bf<<<cdiv(H2 * D_ATT, 256), 256, 0, stream>>>(Wa, Wat, H2, D_ATT);
  k_transpose_bf<<<cdiv(H2 * D_ATT, 256), 256, 0, stream>>>(Wb, Wbt, H2, D_ATT);
  k_f32_to_bf16<<<cdiv(NN * F_IN, 256), 256, 0, stream>>>(x, xbf, NN * F_IN);

  // ---- 2. degree + layer-1 mean aggregation ----
  k_deg<<<cdiv(EE, 256), 256, 0, stream>>>(dstp, deg, EE);
  k_scatter4<<<cdiv(EE * (F_IN / 4), 256), 256, 0, stream>>>(
      x, src, dstp, agg1, EE, F_IN / 4);
  k_mean_bf<<<cdiv(NN * F_IN, 256), 256, 0, stream>>>(agg1, deg, m1bf,
                                                      NN * F_IN, F_IN);

  // ---- 3. SAGE1: h1 = tanh(mean@W1l + b1 + x@W1r)  [N,256] ----
  k_gemm<<<cdiv((MPAD / 64) * (H1 / 64), 8), 256, 0, stream>>>(
      m1bf, W1lt, xbf, W1rt, b1, h1f, h1bf, MPAD, F_IN, H1, 1);

  // ---- 4. layer-2 mean aggregation ----
  k_scatter4<<<cdiv(EE * (H1 / 4), 256), 256, 0, stream>>>(
      h1f, src, dstp, agg2, EE, H1 / 4);
  k_mean_bf<<<cdiv(NN * H1, 256), 256, 0, stream>>>(agg2, deg, m2bf,
                                                    NN * H1, H1);

  // ---- 5. SAGE2: h2 = mean2@W2l + b2 + h1@W2r  [N,512] (bf16 only) ----
  k_gemm<<<cdiv((MPAD / 64) * (H2 / 64), 8), 256, 0, stream>>>(
      m2bf, W2lt, h1bf, W2rt, b2, (float*)nullptr, h2bf, MPAD, H1, H2, 0);

  // ---- 6. fc: h = relu(h2@Wfc + bfc)  [N,512] f32+bf16 ----
  k_gemm<<<cdiv((MPAD / 64) * (H2 / 64), 8), 256, 0, stream>>>(
      h2bf, Wfct, (const __bf16*)nullptr, (const __bf16*)nullptr, bfc, hF, hbf,
      MPAD, H2, H2, 2);

  // ---- 7. gated attention scores -> Aatt[N] ----
  k_attn<<<cdiv((MPAD / 64) * (D_ATT / 16), 8), 256, 0, stream>>>(
      hbf, Wat, Wbt, ba, bb, Wc, Aatt, MPAD);

  // ---- 8. softmax over N; write A_raw into d_out[5:], weights to wbuf ----
  k_softmax<<<1, 1024, 0, stream>>>(Aatt, bc, out + 5, wbuf, NN);

  // ---- 9. attention pooling M = w @ h ----
  k_pool<<<cdiv(NN, 200), 256, 0, stream>>>(hF, wbuf, Mvec, NN);

  // ---- 10. classifier head ----
  k_final<<<1, 256, 0, stream>>>(Mvec, Wcls, bcls, out);
}
